// GroupedQueryAttention_87969520157161
// MI455X (gfx1250) — matrix operand order repaired and checked by reading, not compiled, not run on hardware
//
#include <hip/hip_runtime.h>
#include <hip/hip_bf16.h>
#include <math.h>

// ---------------- problem constants ----------------
#define EMBED   2048
#define NHEADS  32
#define NKV     8
#define HD      64
#define BATCH_N 2
#define SEQ_N   2048
#define KVD     (NKV * HD)          // 512
#define TOK     (BATCH_N * SEQ_N)   // 4096

// ---------------- vector types ----------------
typedef __attribute__((ext_vector_type(16))) __bf16         bf16x16;
typedef __attribute__((ext_vector_type(8)))  float          floatx8;
typedef __attribute__((ext_vector_type(8)))  unsigned short ushortx8;
typedef __attribute__((ext_vector_type(16))) unsigned short ushortx16;

__device__ __forceinline__ unsigned short f32_to_bf16_rne(float f) {
  union { float f; unsigned int u; } v; v.f = f;
  unsigned int r = v.u + 0x7FFFu + ((v.u >> 16) & 1u);
  return (unsigned short)(r >> 16);
}

__device__ __forceinline__ bf16x16 pack_frag(ushortx8 lo, ushortx8 hi) {
  union { ushortx16 u; bf16x16 b; } c;
#pragma unroll
  for (int i = 0; i < 8; ++i) { c.u[i] = lo[i]; c.u[8 + i] = hi[i]; }
  return c.b;
}

// A fragment: 16x32 bf16 tile from row-major storage (leading dim lda, in elems).
// p points at tile origin (row0, k0).  ISA layout:
//   lane L: row = L%16 ; elems 0..7 -> k = (L/16)*8 + e ; elems 8..15 -> +16.
__device__ __forceinline__ bf16x16 load_a_frag(const unsigned short* p, int lda, int lane) {
  const unsigned short* q = p + (lane & 15) * lda + ((lane >> 4) << 3);
  ushortx8 lo = *reinterpret_cast<const ushortx8*>(q);
  ushortx8 hi = *reinterpret_cast<const ushortx8*>(q + 16);
  return pack_frag(lo, hi);
}

// B fragment: 32x16 bf16 tile taken from "B^T" row-major storage BT[n][k]
// (leading dim ldb). p points at (n0, k0).  ISA layout:
//   lane L: col = L%16 ; elems 0..7 -> k = (L/16)*8 + e ; elems 8..15 -> +16.
__device__ __forceinline__ bf16x16 load_b_frag(const unsigned short* p, int ldb, int lane) {
  const unsigned short* q = p + (lane & 15) * ldb + ((lane >> 4) << 3);
  ushortx8 lo = *reinterpret_cast<const ushortx8*>(q);
  ushortx8 hi = *reinterpret_cast<const ushortx8*>(q + 16);
  return pack_frag(lo, hi);
}

__device__ __forceinline__ floatx8 wmma_bf16f32(bf16x16 a, bf16x16 b, floatx8 c) {
  return __builtin_amdgcn_wmma_f32_16x16x32_bf16(false, a, false, b, (short)0, c,
                                                 false, false);
}

// ---------------- fp32 -> bf16 converters ----------------
__global__ void cvt_f32_bf16(const float* __restrict__ src,
                             unsigned short* __restrict__ dst, int n) {
  int i = blockIdx.x * blockDim.x + threadIdx.x;
  if (i < n) dst[i] = f32_to_bf16_rne(src[i]);
}

// WT[n*K + k] = bf16(W[k*N + n])   (weights stored [in,out] -> bf16 [out,in])
__global__ void transpose_f32_bf16(const float* __restrict__ W,
                                   unsigned short* __restrict__ WT, int K, int N) {
  int i = blockIdx.x * blockDim.x + threadIdx.x;
  if (i < K * N) {
    int n = i / K, k = i - n * K;
    WT[i] = f32_to_bf16_rne(W[k * N + n]);
  }
}

// ---------------- generic bf16 WMMA GEMM ----------------
// C[m][n] = sum_k A[m][k] * BT[n][k] + bias[n]
// mode 0: store bf16 row-major (ldc)          (Q, K projections)
// mode 1: store fp32 row-major (ldc)          (final output projection)
// mode 2: store bf16 V-transposed  Vt[((b*NKV+kvh)*HD+d)*SEQ + s]
// Each wave computes a 32x64 tile (2x4 WMMA tiles), K-loop step 32:
// 8 WMMAs per K-step against 12 b128 loads -> WMMA-issue-bound, not VMEM-bound.
__global__ void gemm_bf16_wmma(const unsigned short* __restrict__ A, int lda,
                               const unsigned short* __restrict__ BT, int ldb,
                               const float* __restrict__ bias,
                               unsigned short* __restrict__ outb,
                               float* __restrict__ outf,
                               int M, int N, int K, int mode, int ldc) {
  const int lane = threadIdx.x & 31;
  const int w    = (blockIdx.x * blockDim.x + threadIdx.x) >> 5;
  const int tilesN = N >> 6;
  const int m0 = (w / tilesN) << 5;
  const int n0 = (w % tilesN) << 6;

  floatx8 acc[2][4];
#pragma unroll
  for (int i = 0; i < 2; ++i)
#pragma unroll
    for (int j = 0; j < 4; ++j) acc[i][j] = (floatx8){};

  const unsigned short* Arow[2];
  const unsigned short* Brow[4];
#pragma unroll
  for (int i = 0; i < 2; ++i) Arow[i] = A + (size_t)(m0 + 16 * i) * lda;
#pragma unroll
  for (int j = 0; j < 4; ++j) Brow[j] = BT + (size_t)(n0 + 16 * j) * ldb;

  for (int k0 = 0; k0 < K; k0 += 32) {
    bf16x16 a0 = load_a_frag(Arow[0] + k0, lda, lane);
    bf16x16 a1 = load_a_frag(Arow[1] + k0, lda, lane);
    if (k0 + 32 < K) {                       // gfx1250 global_prefetch path
      __builtin_prefetch(Arow[0] + k0 + 32, 0, 3);
      __builtin_prefetch(Brow[0] + k0 + 32, 0, 3);
      __builtin_prefetch(Brow[2] + k0 + 32, 0, 3);
    }
#pragma unroll
    for (int j = 0; j < 4; ++j) {
      bf16x16 b = load_b_frag(Brow[j] + k0, ldb, lane);
      acc[0][j] = wmma_bf16f32(a0, b, acc[0][j]);
      acc[1][j] = wmma_bf16f32(a1, b, acc[1][j]);
    }
  }

  // D layout: lane L, vgpr r -> row = r + 8*(L/16), col = L%16
  const int rbase = (lane >> 4) << 3;
  const int col   = lane & 15;
#pragma unroll
  for (int i = 0; i < 2; ++i) {
#pragma unroll
    for (int j = 0; j < 4; ++j) {
#pragma unroll
      for (int r = 0; r < 8; ++r) {
        int m = m0 + i * 16 + rbase + r;
        int n = n0 + j * 16 + col;
        float v = acc[i][j][r] + bias[n];
        if (mode == 0) {
          outb[(size_t)m * ldc + n] = f32_to_bf16_rne(v);
        } else if (mode == 1) {
          outf[(size_t)m * ldc + n] = v;
        } else {
          int b = m >> 11, s = m & (SEQ_N - 1);
          int kvh = n >> 6, d = n & (HD - 1);
          outb[(size_t)((b * NKV + kvh) * HD + d) * SEQ_N + s] = f32_to_bf16_rne(v);
        }
      }
    }
  }
}

// ---------------- flash GQA attention ----------------
// One wave handles 16 query rows of one (batch, head) across full T.
// 4 waves per workgroup (disjoint LDS slabs, no barriers needed).
__global__ void gqa_attention(const unsigned short* __restrict__ Qb,   // [TOK, EMBED] bf16
                              const unsigned short* __restrict__ Kb,   // [TOK, KVD]   bf16
                              const unsigned short* __restrict__ Vt,   // [B,NKV,HD,SEQ] bf16
                              unsigned short* __restrict__ Ob) {       // [TOK, EMBED] bf16
  __shared__ float s_scores[4][16 * 32];
  __shared__ float s_alpha[4][16];
  __shared__ float s_l[4][16];
  __shared__ __attribute__((aligned(16))) unsigned short s_p[4][16 * 32];

  const int lane = threadIdx.x & 31;
  const int wid  = threadIdx.x >> 5;
  const int w    = blockIdx.x * 4 + wid;

  const int qblk = w & 127;            // SEQ/16 = 128 query blocks
  const int h    = (w >> 7) & 31;
  const int b    = w >> 12;
  const int kvh  = h >> 2;             // GROUP = 4
  const int tok0 = qblk << 4;

  const unsigned short* Qbase = Qb + (size_t)(b * SEQ_N + tok0) * EMBED + h * HD;
  const unsigned short* Kbase = Kb + (size_t)b * SEQ_N * KVD + kvh * HD;
  const unsigned short* Vbase = Vt + (size_t)((b * NKV + kvh) * HD) * SEQ_N;

  // Q fragments for head_dim 64 = two K-steps of 32, kept resident
  bf16x16 aQ0 = load_a_frag(Qbase + 0,  EMBED, lane);
  bf16x16 aQ1 = load_a_frag(Qbase + 32, EMBED, lane);

  floatx8 o0 = {}, o1 = {}, o2 = {}, o3 = {};
  const int rbase = (lane >> 4) << 3;
  const int col   = lane & 15;
  const int myrow = lane & 15;         // softmax: lane owns row (duplicated per half)
  float m_i = -INFINITY, l_i = 0.0f;
  const float scale = 0.125f;          // 1/sqrt(64)

  for (int t0 = 0; t0 < SEQ_N; t0 += 32) {
    // ---- scores = Q @ K^T (16 x 32), fp32 accum ----
    floatx8 sA = {}, sB = {};
    sA = wmma_bf16f32(aQ0, load_b_frag(Kbase + (size_t)(t0)      * KVD + 0,  KVD, lane), sA);
    sA = wmma_bf16f32(aQ1, load_b_frag(Kbase + (size_t)(t0)      * KVD + 32, KVD, lane), sA);
    sB = wmma_bf16f32(aQ0, load_b_frag(Kbase + (size_t)(t0 + 16) * KVD + 0,  KVD, lane), sB);
    sB = wmma_bf16f32(aQ1, load_b_frag(Kbase + (size_t)(t0 + 16) * KVD + 32, KVD, lane), sB);

#pragma unroll
    for (int r = 0; r < 8; ++r) {
      int row = rbase + r;
      s_scores[wid][row * 32 + col]      = sA[r] * scale;
      s_scores[wid][row * 32 + 16 + col] = sB[r] * scale;
    }

    // ---- online softmax, one lane per row (fp32) ----
    float mblk = -INFINITY;
#pragma unroll
    for (int j = 0; j < 32; ++j) {
      float v = s_scores[wid][myrow * 32 + j];
      mblk = fmaxf(mblk, v);
    }
    float mnew  = fmaxf(m_i, mblk);
    float alpha = __expf(m_i - mnew);
    float sum   = 0.0f;
#pragma unroll
    for (int j = 0; j < 32; ++j) {
      float p = __expf(s_scores[wid][myrow * 32 + j] - mnew);
      sum += p;
      if (lane < 16) s_p[wid][myrow * 32 + j] = f32_to_bf16_rne(p);
    }
    l_i = l_i * alpha + sum;
    m_i = mnew;
    if (lane < 16) s_alpha[wid][myrow] = alpha;

    // ---- rescale accumulators by alpha[row] ----
#pragma unroll
    for (int r = 0; r < 8; ++r) {
      float a = s_alpha[wid][rbase + r];
      o0[r] *= a; o1[r] *= a; o2[r] *= a; o3[r] *= a;
    }

    // ---- O += P @ V  (P from LDS in A-frag layout, V^T B-frags) ----
    bf16x16 aP = load_a_frag(&s_p[wid][0], 32, lane);
    o0 = wmma_bf16f32(aP, load_b_frag(Vbase + (size_t)0  * SEQ_N + t0, SEQ_N, lane), o0);
    o1 = wmma_bf16f32(aP, load_b_frag(Vbase + (size_t)16 * SEQ_N + t0, SEQ_N, lane), o1);
    o2 = wmma_bf16f32(aP, load_b_frag(Vbase + (size_t)32 * SEQ_N + t0, SEQ_N, lane), o2);
    o3 = wmma_bf16f32(aP, load_b_frag(Vbase + (size_t)48 * SEQ_N + t0, SEQ_N, lane), o3);
  }

  // ---- finalize: divide by l[row], store bf16 ----
  if (lane < 16) s_l[wid][myrow] = l_i;
  unsigned short* Obase = Ob + (size_t)(b * SEQ_N + tok0) * EMBED + h * HD;
#pragma unroll
  for (int r = 0; r < 8; ++r) {
    float inv = 1.0f / s_l[wid][rbase + r];
    size_t rowoff = (size_t)(rbase + r) * EMBED;
    Obase[rowoff + 0  + col] = f32_to_bf16_rne(o0[r] * inv);
    Obase[rowoff + 16 + col] = f32_to_bf16_rne(o1[r] * inv);
    Obase[rowoff + 32 + col] = f32_to_bf16_rne(o2[r] * inv);
    Obase[rowoff + 48 + col] = f32_to_bf16_rne(o3[r] * inv);
  }
}

// ---------------- host orchestration ----------------
extern "C" void kernel_launch(void* const* d_in, const int* in_sizes, int n_in,
                              void* d_out, int out_size, void* d_ws, size_t ws_size,
                              hipStream_t stream) {
  const float* x  = (const float*)d_in[0];
  const float* Wq = (const float*)d_in[1];
  const float* bq = (const float*)d_in[2];
  const float* Wk = (const float*)d_in[3];
  const float* bk = (const float*)d_in[4];
  const float* Wv = (const float*)d_in[5];
  const float* bv = (const float*)d_in[6];
  const float* Wo = (const float*)d_in[7];
  const float* bo = (const float*)d_in[8];
  float* out = (float*)d_out;

  size_t off = 0;
  auto salloc = [&](size_t bytes) {
    void* p = (char*)d_ws + off;
    off += (bytes + 255) & ~(size_t)255;
    return p;
  };
  unsigned short* xb  = (unsigned short*)salloc((size_t)TOK * EMBED * 2);
  unsigned short* WqT = (unsigned short*)salloc((size_t)EMBED * EMBED * 2);
  unsigned short* WkT = (unsigned short*)salloc((size_t)KVD * EMBED * 2);
  unsigned short* WvT = (unsigned short*)salloc((size_t)KVD * EMBED * 2);
  unsigned short* WoT = (unsigned short*)salloc((size_t)EMBED * EMBED * 2);
  unsigned short* Qb  = (unsigned short*)salloc((size_t)TOK * EMBED * 2);
  unsigned short* Kb  = (unsigned short*)salloc((size_t)TOK * KVD * 2);
  unsigned short* Vt  = (unsigned short*)salloc((size_t)TOK * KVD * 2);
  unsigned short* Ab  = (unsigned short*)salloc((size_t)TOK * EMBED * 2);

  // fp32 -> bf16 (x) and bf16 weight transposes
  {
    int n = TOK * EMBED;
    cvt_f32_bf16<<<n / 256, 256, 0, stream>>>(x, xb, n);
  }
  transpose_f32_bf16<<<(EMBED * EMBED) / 256, 256, 0, stream>>>(Wq, WqT, EMBED, EMBED);
  transpose_f32_bf16<<<(EMBED * KVD)   / 256, 256, 0, stream>>>(Wk, WkT, EMBED, KVD);
  transpose_f32_bf16<<<(EMBED * KVD)   / 256, 256, 0, stream>>>(Wv, WvT, EMBED, KVD);
  transpose_f32_bf16<<<(EMBED * EMBED) / 256, 256, 0, stream>>>(Wo, WoT, EMBED, EMBED);

  // Q = x @ Wq + bq  -> bf16 [TOK, EMBED]   (waves = 128 * 32 = 4096)
  gemm_bf16_wmma<<<(TOK / 32) * (EMBED / 64) / 8, 256, 0, stream>>>(
      xb, EMBED, WqT, EMBED, bq, Qb, nullptr, TOK, EMBED, EMBED, 0, EMBED);
  // K = x @ Wk + bk  -> bf16 [TOK, KVD]     (waves = 128 * 8 = 1024)
  gemm_bf16_wmma<<<(TOK / 32) * (KVD / 64) / 8, 256, 0, stream>>>(
      xb, EMBED, WkT, EMBED, bk, Kb, nullptr, TOK, KVD, EMBED, 0, KVD);
  // V = x @ Wv + bv  -> bf16 transposed [B, NKV, HD, SEQ]
  gemm_bf16_wmma<<<(TOK / 32) * (KVD / 64) / 8, 256, 0, stream>>>(
      xb, EMBED, WvT, EMBED, bv, Vt, nullptr, TOK, KVD, EMBED, 2, KVD);

  // attention: one wave per 16 query rows per (b, h); 4 waves per block
  gqa_attention<<<(BATCH_N * NHEADS * (SEQ_N / 16)) / 4, 128, 0, stream>>>(Qb, Kb, Vt, Ab);

  // out = attn @ Wo + bo  -> fp32 [TOK, EMBED]
  gemm_bf16_wmma<<<(TOK / 32) * (EMBED / 64) / 8, 256, 0, stream>>>(
      Ab, EMBED, WoT, EMBED, bo, nullptr, out, TOK, EMBED, EMBED, 1, EMBED);

  (void)in_sizes; (void)n_in; (void)out_size; (void)ws_size;
}
